// GlobalModel_TWO_20839181320245
// MI455X (gfx1250) — compile-verified, matrix operand order checked by reading, not run on hardware
//
#include <hip/hip_runtime.h>
#include <cstdint>

#define N_NODES 10000
#define N_EDGES 640000
#define N_GRAPHS 64
#define D_NODE 128
#define D_EDGE 32
#define D_IN 160      // K = D_NODE + D_EDGE
#define HIDDEN 128
#define D_OUT 64
#define EPSF 1e-5f
#define TILE_E 128
#define NTILES (N_EDGES / TILE_E)   // 5000 exactly
#define EDGE_BLOCKS 512

typedef __attribute__((ext_vector_type(4)))  _Float16 v4h;
typedef __attribute__((ext_vector_type(8)))  _Float16 v8h;
typedef __attribute__((ext_vector_type(16))) _Float16 v16h;
typedef __attribute__((ext_vector_type(8)))  float    v8f;

extern __shared__ char smem[];

// Low 32 bits of a generic pointer into LDS == the wave-relative LDS byte offset
// (ISA 10.2).
__device__ __forceinline__ unsigned lds_off(const void* p) {
  return (unsigned)(unsigned long long)p;
}

// gfx1250 async copy: 16B global -> LDS, per-lane addressed, tracked by ASYNCcnt.
__device__ __forceinline__ void async_copy_b128(unsigned lds_byte_off, const void* gsrc) {
  asm volatile("global_load_async_to_lds_b128 %0, %1, off"
               :: "v"(lds_byte_off), "v"(gsrc) : "memory");
}

__device__ __forceinline__ void atomic_add_f32_agent(float* p, float v) {
  __hip_atomic_fetch_add(p, v, __ATOMIC_RELAXED, __HIP_MEMORY_SCOPE_AGENT);
}

// ---------------------------------------------------------------------------
// Prep: x -> f16, W1 -> f16 transposed [n][k]; zero global accumulators.
// ---------------------------------------------------------------------------
__global__ __launch_bounds__(256) void prep_kernel(
    const float* __restrict__ x, const float* __restrict__ W1,
    _Float16* __restrict__ xh, _Float16* __restrict__ w1t,
    float* __restrict__ Sg, float* __restrict__ Sq, unsigned* __restrict__ Cnt) {
  const int stride = gridDim.x * blockDim.x;
  const int gid = blockIdx.x * blockDim.x + threadIdx.x;
  for (int i = gid; i < N_NODES * D_NODE; i += stride) xh[i] = (_Float16)x[i];
  for (int i = gid; i < D_IN * HIDDEN; i += stride) {
    int k = i / HIDDEN, n = i % HIDDEN;
    w1t[n * D_IN + k] = (_Float16)W1[i];
  }
  for (int i = gid; i < N_GRAPHS * HIDDEN; i += stride) Sg[i] = 0.f;
  for (int i = gid; i < HIDDEN; i += stride) Sq[i] = 0.f;
  for (int i = gid; i < N_GRAPHS; i += stride) Cnt[i] = 0u;
}

// ---------------------------------------------------------------------------
// Edge kernel: a_e = leaky(concat(x[row],eattr) @ W1 + b1) via f16 WMMA.
// Wave w owns output N-tile w (B fragments live in registers for the whole
// kernel); A tiles are double-buffered in LDS and filled by async copies.
// Accumulates per-graph sums S[64][128], column sum-of-squares, edge counts.
// ---------------------------------------------------------------------------
__global__ __launch_bounds__(256) void edge_kernel(
    const int* __restrict__ ei, const float* __restrict__ eattr,
    const int* __restrict__ batch, const float* __restrict__ b1,
    const _Float16* __restrict__ xh, const _Float16* __restrict__ w1t,
    float* __restrict__ Sg, float* __restrict__ Sq, unsigned* __restrict__ Cnt) {
  _Float16* sA = (_Float16*)smem;                  // 2 x [TILE_E][160] f16
  float* sS = (float*)(sA + 2 * TILE_E * D_IN);    // [64][128] per-graph sums
  unsigned* sCnt = (unsigned*)(sS + N_GRAPHS * HIDDEN);  // [64]
  int* sGid = (int*)(sCnt + N_GRAPHS);             // 2 x [TILE_E]

  const int tid = threadIdx.x;
  for (int i = tid; i < N_GRAPHS * HIDDEN; i += 256) sS[i] = 0.f;
  if (tid < N_GRAPHS) sCnt[tid] = 0u;

  const int wave = tid >> 5;       // 8 waves -> 8 N-tiles of 16 features
  const int lane = tid & 31;
  const int lh   = lane & 15;
  const int hi   = lane >> 4;      // K-half select per 16-bit WMMA layout
  const int n    = wave * 16 + lh; // output feature column owned by this lane
  const float bias = b1[n];

  // B fragments (W1^T rows for column n), register-resident for whole kernel.
  v16h Bfrag[5];
  {
    const char* bp = (const char*)(w1t + (size_t)n * D_IN);
    #pragma unroll
    for (int kk = 0; kk < 5; ++kk) {
      v8h b0 = *(const v8h*)(bp + kk * 64 + hi * 16);
      v8h b1v = *(const v8h*)(bp + kk * 64 + hi * 16 + 32);
      Bfrag[kk] = __builtin_shufflevector(b0, b1v,
                    0,1,2,3,4,5,6,7,8,9,10,11,12,13,14,15);
    }
  }

  const unsigned sA_off = lds_off(sA);
  const int le = tid >> 1;         // 2 threads stage each edge
  const int hf = tid & 1;

  auto stage = [&](int t, int buf) {
    const int e = t * TILE_E + le;
    const int row = ei[e];
    {   // x row: 256 B of f16, 8 async 16B chunks per thread (8 per wave)
      const char* src = (const char*)(xh + (size_t)row * D_NODE) + hf * 128;
      unsigned dst = sA_off + (unsigned)buf * (TILE_E * D_IN * 2)
                   + (unsigned)le * (D_IN * 2) + (unsigned)hf * 128;
      #pragma unroll
      for (int c = 0; c < 8; ++c) async_copy_b128(dst + c * 16, src + c * 16);
    }
    {   // edge_attr: coalesced f32 load, convert, ds_store f16
      const float4* ea = (const float4*)(eattr + (size_t)e * D_EDGE + hf * 16);
      _Float16* dq = sA + buf * (TILE_E * D_IN) + le * D_IN + D_NODE + hf * 16;
      #pragma unroll
      for (int j = 0; j < 4; ++j) {
        float4 v = ea[j];
        v4h hq; hq[0] = (_Float16)v.x; hq[1] = (_Float16)v.y;
                hq[2] = (_Float16)v.z; hq[3] = (_Float16)v.w;
        *(v4h*)(dq + j * 4) = hq;
      }
    }
    if (hf == 0) {
      int g = batch[ei[N_EDGES + e]];
      sGid[buf * TILE_E + le] = g;
      atomicAdd(&sCnt[g], 1u);
    }
  };

  float sqacc = 0.f;               // per-lane sum of a^2 for column n
  stage(blockIdx.x, 0);            // prologue: tile 0 into buffer 0

  int i = 0;
  for (int t = blockIdx.x; t < NTILES; t += EDGE_BLOCKS, ++i) {
    const int cur = i & 1;
    const int tn = t + EDGE_BLOCKS;
    if (tn < NTILES) {
      stage(tn, cur ^ 1);                       // prefetch next tile
      __builtin_amdgcn_s_wait_asynccnt(8);      // current tile's copies done
    } else {
      __builtin_amdgcn_s_wait_asynccnt(0);
    }
    __syncthreads();   // all waves' copies + ds stores for tile t visible

    const _Float16* Abase = sA + cur * (TILE_E * D_IN);
    const int* gbase = sGid + cur * TILE_E;

    #pragma unroll
    for (int mt = 0; mt < 8; ++mt) {
      v8f acc = {0.f,0.f,0.f,0.f,0.f,0.f,0.f,0.f};
      #pragma unroll
      for (int kk = 0; kk < 5; ++kk) {
        const char* ap = (const char*)(Abase + (mt * 16 + lh) * D_IN)
                       + kk * 64 + hi * 16;
        v8h a0 = *(const v8h*)ap;
        v8h a1 = *(const v8h*)(ap + 32);
        v16h A = __builtin_shufflevector(a0, a1,
                   0,1,2,3,4,5,6,7,8,9,10,11,12,13,14,15);
        acc = __builtin_amdgcn_wmma_f32_16x16x32_f16(
            false, A, false, Bfrag[kk], (short)0, acc, false, false);
      }
      // C/D layout: VGPR r, lanes<16 -> M=r, lanes>=16 -> M=r+8
      #pragma unroll
      for (int r = 0; r < 8; ++r) {
        float a = acc[r] + bias;
        a = a > 0.f ? a : 0.01f * a;            // leaky_relu
        sqacc += a * a;
        int g = gbase[mt * 16 + hi * 8 + r];
        atomicAdd(&sS[g * HIDDEN + n], a);      // ds_add_f32
      }
    }
    __syncthreads();   // buffer cur may be re-staged next iteration
  }

  atomic_add_f32_agent(&Sq[n], sqacc);
  for (int i2 = tid; i2 < N_GRAPHS * HIDDEN; i2 += 256)
    atomic_add_f32_agent(&Sg[i2], sS[i2]);
  if (tid < N_GRAPHS) atomicAdd(&Cnt[tid], sCnt[tid]);
}

// ---------------------------------------------------------------------------
// Per-graph pooling of x (batch is sorted -> deterministic segmented sum).
// ---------------------------------------------------------------------------
__device__ __forceinline__ int lower_bound_i(const int* a, int n, int v) {
  int lo = 0, hi = n;
  while (lo < hi) { int m = (lo + hi) >> 1; if (a[m] < v) lo = m + 1; else hi = m; }
  return lo;
}

__global__ __launch_bounds__(256) void gx_kernel(
    const float* __restrict__ x, const int* __restrict__ batch,
    float* __restrict__ GX) {
  __shared__ float red[256];
  const int g = blockIdx.x;
  const int start = lower_bound_i(batch, N_NODES, g);
  const int end   = lower_bound_i(batch, N_NODES, g + 1);
  const int f = threadIdx.x & 127;
  const int h = threadIdx.x >> 7;
  float s = 0.f;
  for (int i = start + h; i < end; i += 2) s += x[(size_t)i * D_NODE + f];
  red[threadIdx.x] = s;
  __syncthreads();
  if (h == 0) GX[g * D_NODE + f] = red[f] + red[128 + f];
}

// ---------------------------------------------------------------------------
// Final: BN1 stats -> Z -> go = Z@W2 + cnt*b2 -> MLP2 (fp32, tiny).
// ---------------------------------------------------------------------------
__global__ __launch_bounds__(256) void final_kernel(
    const float* __restrict__ Sg, const float* __restrict__ Sq,
    const unsigned* __restrict__ Cnt, const float* __restrict__ GX,
    const float* __restrict__ g1, const float* __restrict__ be1,
    const float* __restrict__ W2, const float* __restrict__ b2,
    const float* __restrict__ W3, const float* __restrict__ b3,
    const float* __restrict__ g3, const float* __restrict__ be3,
    const float* __restrict__ W4, const float* __restrict__ b4,
    float* __restrict__ out) {
  float* sZ  = (float*)smem;              // [64][128], later reused for t
  float* sGo = sZ + N_GRAPHS * HIDDEN;    // [64][128]
  float* sMu = sGo + N_GRAPHS * HIDDEN;   // [128]
  float* sSc = sMu + HIDDEN;              // [128]
  float* sC  = sSc + HIDDEN;              // [64]
  const int tid = threadIdx.x;

  if (tid < N_GRAPHS) sC[tid] = (float)Cnt[tid];
  if (tid < HIDDEN) {
    float cs = 0.f;
    for (int g = 0; g < N_GRAPHS; ++g) cs += Sg[g * HIDDEN + tid];
    float mu = cs / (float)N_EDGES;
    float var = Sq[tid] / (float)N_EDGES - mu * mu;   // biased var (jnp.var)
    sMu[tid] = mu;
    sSc[tid] = g1[tid] * rsqrtf(var + EPSF);
  }
  __syncthreads();

  // Z[g] = sum over edges of BN1(a_e) (affine identity on group sums)
  for (int i = tid; i < N_GRAPHS * HIDDEN; i += 256) {
    int g = i >> 7, n = i & 127;
    float c = sC[g];
    sZ[i] = sSc[n] * (Sg[i] - c * sMu[n]) + c * be1[n];
  }
  __syncthreads();

  // go = Z @ W2 + cnt * b2
  for (int i = tid; i < N_GRAPHS * HIDDEN; i += 256) {
    int g = i >> 7, n = i & 127;
    float acc = sC[g] * b2[n];
    for (int k = 0; k < HIDDEN; ++k) acc += sZ[g * HIDDEN + k] * W2[k * HIDDEN + n];
    sGo[i] = acc;
  }
  __syncthreads();

  // t = leaky( [GX | go] @ W3 + b3 )
  for (int i = tid; i < N_GRAPHS * HIDDEN; i += 256) {
    int g = i >> 7, n = i & 127;
    float acc = b3[n];
    for (int k = 0; k < D_NODE; ++k)  acc += GX[g * D_NODE + k] * W3[k * HIDDEN + n];
    for (int k = 0; k < HIDDEN; ++k)  acc += sGo[g * HIDDEN + k] * W3[(D_NODE + k) * HIDDEN + n];
    sZ[i] = acc > 0.f ? acc : 0.01f * acc;
  }
  __syncthreads();

  // BN over 64 rows
  if (tid < HIDDEN) {
    float m = 0.f;
    for (int g = 0; g < N_GRAPHS; ++g) m += sZ[g * HIDDEN + tid];
    m /= (float)N_GRAPHS;
    float v = 0.f;
    for (int g = 0; g < N_GRAPHS; ++g) { float d = sZ[g * HIDDEN + tid] - m; v += d * d; }
    v /= (float)N_GRAPHS;
    sMu[tid] = m;
    sSc[tid] = g3[tid] * rsqrtf(v + EPSF);
  }
  __syncthreads();
  for (int i = tid; i < N_GRAPHS * HIDDEN; i += 256) {
    int n = i & 127;
    sZ[i] = sSc[n] * (sZ[i] - sMu[n]) + be3[n];
  }
  __syncthreads();

  // out = t @ W4 + b4
  for (int i = tid; i < N_GRAPHS * D_OUT; i += 256) {
    int g = i >> 6, o = i & 63;
    float acc = b4[o];
    for (int n = 0; n < HIDDEN; ++n) acc += sZ[g * HIDDEN + n] * W4[n * D_OUT + o];
    out[i] = acc;
  }
}

// ---------------------------------------------------------------------------
extern "C" void kernel_launch(void* const* d_in, const int* in_sizes, int n_in,
                              void* d_out, int out_size, void* d_ws, size_t ws_size,
                              hipStream_t stream) {
  const float* x     = (const float*)d_in[0];
  const int*   ei    = (const int*)  d_in[1];
  const float* eattr = (const float*)d_in[2];
  // d_in[3] = u (unused by the reference)
  const int*   batch = (const int*)  d_in[4];
  const float* W1  = (const float*)d_in[5];
  const float* b1  = (const float*)d_in[6];
  const float* g1  = (const float*)d_in[7];
  const float* be1 = (const float*)d_in[8];
  const float* W2  = (const float*)d_in[9];
  const float* b2  = (const float*)d_in[10];
  const float* W3  = (const float*)d_in[11];
  const float* b3  = (const float*)d_in[12];
  const float* g3  = (const float*)d_in[13];
  const float* be3 = (const float*)d_in[14];
  const float* W4  = (const float*)d_in[15];
  const float* b4  = (const float*)d_in[16];

  char* w = (char*)d_ws;
  _Float16* xh  = (_Float16*)w; w += (size_t)N_NODES * D_NODE * 2;   // 2,560,000 B
  _Float16* w1t = (_Float16*)w; w += (size_t)D_IN * HIDDEN * 2;      //    40,960 B
  float*    Sgp = (float*)w;    w += (size_t)N_GRAPHS * HIDDEN * 4;  //    32,768 B
  float*    Sqp = (float*)w;    w += (size_t)HIDDEN * 4;             //       512 B
  unsigned* Cnt = (unsigned*)w; w += 256;                            //       256 B
  float*    GX  = (float*)w;    w += (size_t)N_GRAPHS * HIDDEN * 4;  //    32,768 B

  prep_kernel<<<1024, 256, 0, stream>>>(x, W1, xh, w1t, Sgp, Sqp, Cnt);

  size_t smem_edge = (size_t)2 * TILE_E * D_IN * 2   // sA x2   81,920
                   + (size_t)N_GRAPHS * HIDDEN * 4   // sS      32,768
                   + (size_t)N_GRAPHS * 4            // sCnt       256
                   + (size_t)2 * TILE_E * 4;         // sGid x2  1,024 => 115,968
  edge_kernel<<<EDGE_BLOCKS, 256, smem_edge, stream>>>(ei, eattr, batch, b1,
                                                       xh, w1t, Sgp, Sqp, Cnt);

  gx_kernel<<<N_GRAPHS, 256, 0, stream>>>(x, batch, GX);

  size_t smem_fin = (size_t)(2 * N_GRAPHS * HIDDEN + 2 * HIDDEN + N_GRAPHS) * 4;
  final_kernel<<<1, 256, smem_fin, stream>>>(Sgp, Sqp, Cnt, GX, g1, be1,
                                             W2, b2, W3, b3, g3, be3, W4, b4,
                                             (float*)d_out);
}
